// SINKT_58686433133100
// MI455X (gfx1250) — matrix operand order repaired and checked by reading, not compiled, not run on hardware
//
// SINKT forward for MI455X (gfx1250): WMMA-bf16 GEMMs with async-to-LDS A-panel
// staging, atomic GAT softmax, persistent-LDS WMMA GRU. Compile-only target;
// d_in flattening assumes top-level insertion order with `params` flattened by
// sorted keys (JAX pytree).
#include <hip/hip_runtime.h>
#include <math.h>

#define NQ   50000
#define NC   1000
#define NCP  1008      // NC padded to multiple of 16
#define BERT 768
#define DD   128
#define BSB  64
#define LLL  200
#define BL   (BSB*LLL) // 12800
#define HH   256       // GRU hidden 2*D
#define GG   768       // 3*HH
#define ECCN 20000
#define EQCN 150000
#define ECQN 150000

typedef __attribute__((ext_vector_type(16))) __bf16 v16bf;
typedef __attribute__((ext_vector_type(8)))  float  v8f;
typedef unsigned short u16;
typedef unsigned int   u32;

#define CDIV(a,b) (((a)+(b)-1)/(b))

// ---------------- helpers ----------------
__device__ __forceinline__ u16 f2bf(float f) {
  u32 u = __float_as_uint(f);
  u += 0x7FFFu + ((u >> 16) & 1u);   // round-to-nearest-even
  return (u16)(u >> 16);
}
__device__ __forceinline__ float bf2f(u16 h) {
  return __uint_as_float(((u32)h) << 16);
}
// orderable-uint encoding for float atomicMax
__device__ __forceinline__ u32 f32_ord(float x) {
  u32 u = __float_as_uint(x);
  return (u & 0x80000000u) ? ~u : (u | 0x80000000u);
}
__device__ __forceinline__ float ord_f32(u32 u) {
  return __uint_as_float((u & 0x80000000u) ? (u & 0x7FFFFFFFu) : ~u);
}

// ---------------- elementwise ----------------
__global__ void k_fill_u32(u32* p, u32 v, int n) {
  int i = blockIdx.x * 256 + threadIdx.x;
  if (i < n) p[i] = v;
}
__global__ void k_f2bf(u16* d, const float* s, int n) {
  int i = blockIdx.x * 256 + threadIdx.x;
  if (i < n) d[i] = f2bf(s[i]);
}

// ---------------- WMMA GEMM:  C[M x N] = A[M x K] * W[N x K]^T ----------------
// One block (4 wave32s) owns one 16-row M-tile. The 16xK A panel (contiguous
// 32*K bytes) is staged into LDS with global_load_async_to_lds_b128 (ASYNCcnt),
// then each wave sweeps 64-column N-strips: A fragments come from LDS as two
// ds_load_b128 per K-step, B fragments stream from global (torch [N x K]
// weight layout -> per-lane 32B contiguous runs).
__global__ __launch_bounds__(128)
void k_gemm_wmma(const u16* __restrict__ A, const u16* __restrict__ B,
                 float* __restrict__ C, int Mt, int Ng, int K) {
  (void)Mt;
  extern __shared__ u16 aS[];                 // [16 x K] bf16
  int tid = threadIdx.x, lane = tid & 31, w = tid >> 5;
  int mt = blockIdx.x;
  // ---- async stage: flat copy of the A panel into LDS ----
  {
    const char* gsrc = (const char*)(A + (size_t)mt * 16 * K);
    u32 lbase = (u32)(size_t)(void*)aS;       // LDS byte offset (addr[31:0])
    for (int i = tid; i < 2 * K; i += 128) {  // 2*K chunks of 16 B
      u32 loff = lbase + (u32)i * 16u;
      unsigned long long ga = (unsigned long long)(size_t)(gsrc + (size_t)i * 16);
      asm volatile("global_load_async_to_lds_b128 %0, %1, off"
                   :: "v"(loff), "v"(ga) : "memory");
    }
    asm volatile("s_wait_asynccnt 0" ::: "memory");
  }
  __syncthreads();

  int ml = lane & 15, kh = lane >> 4;
  const u16* arow = aS + ml * K + kh * 16;
  for (int ng = w; ng < Ng; ng += 4) {
    int n0 = ng * 64 + ml;
    const u16* b0 = B + (size_t)(n0)      * K + kh * 16;
    const u16* b1 = B + (size_t)(n0 + 16) * K + kh * 16;
    const u16* b2 = B + (size_t)(n0 + 32) * K + kh * 16;
    const u16* b3 = B + (size_t)(n0 + 48) * K + kh * 16;
    v8f acc0 = {}, acc1 = {}, acc2 = {}, acc3 = {};
    for (int k0 = 0; k0 < K; k0 += 32) {
      union { uint4 q[2]; v16bf v; } a;
      a.q[0] = *(const uint4*)(arow + k0);        // K = k0 + kh*16 + 0..7
      a.q[1] = *(const uint4*)(arow + k0 + 8);    // K = k0 + kh*16 + 8..15
      union { uint4 q[2]; v16bf v; } bb;
      bb.q[0] = *(const uint4*)(b0 + k0); bb.q[1] = *(const uint4*)(b0 + k0 + 8);
      acc0 = __builtin_amdgcn_wmma_f32_16x16x32_bf16(false, a.v, false, bb.v, (short)0, acc0, false, false);
      bb.q[0] = *(const uint4*)(b1 + k0); bb.q[1] = *(const uint4*)(b1 + k0 + 8);
      acc1 = __builtin_amdgcn_wmma_f32_16x16x32_bf16(false, a.v, false, bb.v, (short)0, acc1, false, false);
      bb.q[0] = *(const uint4*)(b2 + k0); bb.q[1] = *(const uint4*)(b2 + k0 + 8);
      acc2 = __builtin_amdgcn_wmma_f32_16x16x32_bf16(false, a.v, false, bb.v, (short)0, acc2, false, false);
      bb.q[0] = *(const uint4*)(b3 + k0); bb.q[1] = *(const uint4*)(b3 + k0 + 8);
      acc3 = __builtin_amdgcn_wmma_f32_16x16x32_bf16(false, a.v, false, bb.v, (short)0, acc3, false, false);
    }
    int ldc = Ng * 64;
    int nb = ng * 64 + ml, mb = mt * 16 + kh * 8;
#pragma unroll
    for (int r = 0; r < 8; ++r) {
      float* c = C + (size_t)(mb + r) * ldc + nb;
      c[0] = acc0[r]; c[16] = acc1[r]; c[32] = acc2[r]; c[48] = acc3[r];
    }
  }
}

// ---------------- score kernels ----------------
// out[i] = dot(X[i, 0:128], v)    (X already offset to column block, ld stride)
__global__ void k_rowdot128(const float* __restrict__ X, int ld,
                            const float* __restrict__ v, float* out, int n) {
  int wid = (blockIdx.x * 256 + threadIdx.x) >> 5, lane = threadIdx.x & 31;
  if (wid >= n) return;
  const float* x = X + (long)wid * ld;
  float s = 0.f;
  for (int j = lane; j < DD; j += 32) s += x[j] * v[j];
  for (int o = 16; o; o >>= 1) s += __shfl_xor(s, o, 32);
  if (!lane) out[wid] = s;
}
// out[i] = dot(X[i, 0:din], v)
__global__ void k_matvec(const float* __restrict__ X, int din,
                         const float* __restrict__ v, float* out, int n) {
  int wid = (blockIdx.x * 256 + threadIdx.x) >> 5, lane = threadIdx.x & 31;
  if (wid >= n) return;
  const float* x = X + (long)wid * din;
  float s = 0.f;
  for (int j = lane; j < din; j += 32) s += x[j] * v[j];
  for (int o = 16; o; o >>= 1) s += __shfl_xor(s, o, 32);
  if (!lane) out[wid] = s;
}
// veff[k] = sum_n Wd[n][k] * ad[n]   (folds hd@ad into x_dst @ veff)
__global__ void k_veff(const float* __restrict__ Wd, const float* __restrict__ ad,
                       float* veff, int din) {
  int k = blockIdx.x * 256 + threadIdx.x;
  if (k >= din) return;
  float s = 0.f;
  for (int n = 0; n < DD; ++n) s += Wd[(long)n * din + k] * ad[n];
  veff[k] = s;
}

// ---------------- GAT edge softmax (3 phases) ----------------
__global__ void k_edge_max(const int* s, const int* d, const float* ss,
                           const float* sd, float* ebuf, u32* menc, int E) {
  int e = blockIdx.x * 256 + threadIdx.x;
  if (e >= E) return;
  float v = ss[s[e]] + sd[d[e]];
  v = v > 0.f ? v : 0.2f * v;                       // leaky_relu(0.2)
  ebuf[e] = v;
  atomicMax(&menc[d[e]], f32_ord(v));
}
__global__ void k_edge_exp(const int* d, float* ebuf, const u32* menc,
                           float* den, int E) {
  int e = blockIdx.x * 256 + threadIdx.x;
  if (e >= E) return;
  int dd = d[e];
  float ex = __expf(ebuf[e] - ord_f32(menc[dd]));
  ebuf[e] = ex;
  atomicAdd(&den[dd], ex);
}
// wave per edge; lanes cover the 128 feature dims
__global__ void k_edge_agg(const int* s, const int* d, const float* ebuf,
                           const float* den, const float* __restrict__ hs, int ldh,
                           float* agg, int E) {
  int gid = blockIdx.x * 256 + threadIdx.x;
  int e = gid >> 5, lane = gid & 31;
  if (e >= E) return;
  float w = ebuf[e] / den[d[e]];
  const float* h = hs + (long)s[e] * ldh;
  float* a = agg + (long)d[e] * DD;
  for (int j = lane; j < DD; j += 32) atomicAdd(&a[j], w * h[j]);
}
__global__ void k_combine_relu(float* out, const float* agg, const float* gemm,
                               int ldg, int coloff, const float* b0,
                               const float* b1, const float* b2, int nrows) {
  int i = blockIdx.x * 256 + threadIdx.x;
  if (i >= nrows * DD) return;
  int r = i >> 7, j = i & 127;
  float v = agg[i] + gemm[(long)r * ldg + coloff + j] + b0[j] + b1[j] +
            (b2 ? b2[j] : 0.f);
  out[i] = v > 0.f ? v : 0.f;
}

// ---------------- sequence phase ----------------
__global__ void k_embed_q(const int* seq, const float* q2, float* eq) {
  int i = blockIdx.x * 256 + threadIdx.x;
  if (i >= BL * DD) return;
  int r = i >> 7, j = i & 127, idx = seq[r];
  eq[i] = (idx > 0) ? q2[(long)(idx - 1) * DD + j] : 0.f;
}
__global__ void k_embed_c(const int* con, const float* c2, float* ec) {
  int i = blockIdx.x * 256 + threadIdx.x;
  if (i >= BL * DD) return;
  int r = i >> 7, j = i & 127;
  float s = 0.f; int cnt = 0;
#pragma unroll
  for (int k = 0; k < 8; ++k) {
    int idx = con[r * 8 + k];
    if (idx > 0) { s += c2[(long)(idx - 1) * DD + j]; ++cnt; }
  }
  ec[i] = s / (float)(cnt ? cnt : 1);
}
__global__ void k_pack_x(const float* ec, const float* op, u16* x) {
  int i = blockIdx.x * 256 + threadIdx.x;
  if (i >= BL * DD) return;
  int r = i >> 7, j = i & 127;
  float o = op[r], v = ec[i];
  x[(long)r * HH + j]      = f2bf(v * o);
  x[(long)r * HH + DD + j] = f2bf(v * (1.f - o));
}
__global__ void k_pack_px(const float* allh, const float* eq, const float* ec,
                          u16* px) {
  int i = blockIdx.x * 256 + threadIdx.x;
  if (i >= BL * 512) return;
  int r = i >> 9, j = i & 511;
  float v = (j < 256) ? allh[(long)r * HH + j]
          : (j < 384) ? eq[(long)r * DD + (j - 256)]
                      : ec[(long)r * DD + (j - 384)];
  px[i] = f2bf(v);
}

// ---------------- persistent GRU: one workgroup, WMMA per step ----------------
// h kept in LDS (bf16, [64][264]); gh = h @ Whh^T in LDS f32 [64][776].
// allh[b*L+t] stores h_{t-1} (pre-shifted for the head).
__global__ __launch_bounds__(512)
void k_gru(const u16* __restrict__ Whh, const float* __restrict__ gi,
           const float* __restrict__ bih, const float* __restrict__ bhh,
           float* __restrict__ allh) {
  extern __shared__ char smem[];
  const int HPAD = 264, GPAD = 776;
  u16*   hS  = (u16*)smem;                     // 64*264*2 = 33792 B
  float* ghS = (float*)(smem + 64 * HPAD * 2); // 64*776*4 = 198656 B
  int tid = threadIdx.x, lane = tid & 31, w = tid >> 5;
  for (int i = tid; i < 64 * HPAD; i += 512) hS[i] = 0;
  for (int i = tid; i < BSB * HH; i += 512) {
    int b = i >> 8, j = i & 255;
    allh[((long)b * LLL) * HH + j] = 0.f;      // h_{-1} = 0
  }
  __syncthreads();
  for (int t = 0; t < LLL; ++t) {
    // 4 M-tiles x 48 N-tiles = 192 tiles; 16 waves * 12 tiles
    for (int idx = 0; idx < 12; ++idx) {
      int tile = w * 12 + idx;
      int mt = tile & 3, nt = tile >> 2;
      int ml = lane & 15, kh = lane >> 4;
      const u16* ha   = hS + (mt * 16 + ml) * HPAD + kh * 16;
      const u16* brow = Whh + (long)(nt * 16 + ml) * HH + kh * 16;
      v8f acc = {};
      for (int k0 = 0; k0 < HH; k0 += 32) {
        union { uint4 q[2]; v16bf v; } a;
        a.q[0] = *(const uint4*)(ha + k0);        // two ds_load_b128
        a.q[1] = *(const uint4*)(ha + k0 + 8);
        union { uint4 q4[2]; v16bf v; } bb;
        bb.q4[0] = *(const uint4*)(brow + k0);
        bb.q4[1] = *(const uint4*)(brow + k0 + 8);
        acc = __builtin_amdgcn_wmma_f32_16x16x32_bf16(false, a.v, false, bb.v,
                                                      (short)0, acc, false, false);
      }
      int nb = nt * 16 + ml, mb = mt * 16 + kh * 8;
#pragma unroll
      for (int r = 0; r < 8; ++r) ghS[(mb + r) * GPAD + nb] = acc[r];
    }
    __syncthreads();
    for (int i = tid; i < BSB * HH; i += 512) {
      int b = i >> 8, j = i & 255;
      long gr = ((long)b * LLL + t) * GG;
      float gir = gi[gr + j]       + bih[j];
      float giz = gi[gr + 256 + j] + bih[256 + j];
      float gin = gi[gr + 512 + j] + bih[512 + j];
      float ghr = ghS[b * GPAD + j]       + bhh[j];
      float ghz = ghS[b * GPAD + 256 + j] + bhh[256 + j];
      float ghn = ghS[b * GPAD + 512 + j] + bhh[512 + j];
      float h   = bf2f(hS[b * HPAD + j]);
      float r = 1.f / (1.f + __expf(-(gir + ghr)));
      float z = 1.f / (1.f + __expf(-(giz + ghz)));
      float n = tanhf(gin + r * ghn);
      float hn = (1.f - z) * n + z * h;
      if (t + 1 < LLL) allh[((long)b * LLL + t + 1) * HH + j] = hn;
      hS[b * HPAD + j] = f2bf(hn);
    }
    __syncthreads();
  }
}

// prob = sigmoid( sum_j tanh(hid[j]+p1b[j]) * p2w[j] + p2b )
__global__ void k_head(const float* __restrict__ hid, const float* p1b,
                       const float* p2w, const float* p2b, float* out, int n) {
  int wid = (blockIdx.x * 256 + threadIdx.x) >> 5, lane = threadIdx.x & 31;
  if (wid >= n) return;
  const float* row = hid + (long)wid * DD;
  float s = 0.f;
  for (int j = lane; j < DD; j += 32) s += tanhf(row[j] + p1b[j]) * p2w[j];
  for (int o = 16; o; o >>= 1) s += __shfl_xor(s, o, 32);
  if (!lane) out[wid] = 1.f / (1.f + __expf(-(s + p2b[0])));
}

// ===================== host side =====================
static void gemm(hipStream_t st, const u16* A, const u16* B, float* C,
                 int Mt, int Ng, int K) {
  k_gemm_wmma<<<Mt, 128, (size_t)32 * K, st>>>(A, B, C, Mt, Ng, K);
}
static void conv(hipStream_t st, u16* d, const float* s, int n) {
  k_f2bf<<<CDIV(n, 256), 256, 0, st>>>(d, s, n);
}
static void fillz(hipStream_t st, void* p, int n_u32) {
  k_fill_u32<<<CDIV(n_u32, 256), 256, 0, st>>>((u32*)p, 0u, n_u32);
}

// One hetero layer. Param order within L[]: 0 ccWd 1 ccWs 2 ccad 3 ccas 4 ccb
// 5 cqWd 6 cqWs 7 cqad 8 cqas 9 cqb 10 lincW 11 lincb 12 linqW 13 linqb
// 14 qcWd 15 qcWs 16 qcad 17 qcas 18 qcb
static void run_layer(hipStream_t st, int din, const float* const* L,
                      const float* qin, const u16* qinbf,
                      const float* cin, const u16* cinbf,
                      const int* ccS, const int* ccD, const int* qcS,
                      const int* qcD, const int* cqS, const int* cqD,
                      u16* qpack, u16* cpack, float* qout, float* cout,
                      float* ssq, float* sdq, float* sscc, float* sscq,
                      float* sdccc, float* sdcqc, u32* menc, float* den,
                      float* ebuf, float* veff, float* aggq, float* aggc,
                      float* q_out, float* c_out) {
  // pack bf16 weights: qout cols [qc.Ws | linq_W], cout cols [cc.Ws | cq.Ws | linc_W]
  conv(st, qpack,             L[15], DD * din);
  conv(st, qpack + DD * din,  L[12], DD * din);
  conv(st, cpack,             L[1],  DD * din);
  conv(st, cpack + DD * din,  L[6],  DD * din);
  conv(st, cpack + 2*DD*din,  L[10], DD * din);
  gemm(st, qinbf, qpack, qout, NQ / 16, 4, din);   // [NQ x 256]
  gemm(st, cinbf, cpack, cout, NCP / 16, 6, din);  // [NCP x 384]
  // source scores (hs @ as) and folded dst scores (x_dst @ (Wd^T ad))
  k_rowdot128<<<CDIV(NQ, 8), 256, 0, st>>>(qout,       256, L[17], ssq,  NQ);
  k_rowdot128<<<CDIV(NC, 8), 256, 0, st>>>(cout,       384, L[3],  sscc, NC);
  k_rowdot128<<<CDIV(NC, 8), 256, 0, st>>>(cout + DD,  384, L[8],  sscq, NC);
  k_veff<<<CDIV(din,256),256,0,st>>>(L[14], L[16], veff, din);
  k_matvec<<<CDIV(NC,8),256,0,st>>>(cin, din, veff, sdcqc, NC);  // qc dst=c
  k_veff<<<CDIV(din,256),256,0,st>>>(L[0], L[2], veff, din);
  k_matvec<<<CDIV(NC,8),256,0,st>>>(cin, din, veff, sdccc, NC);  // cc dst=c
  k_veff<<<CDIV(din,256),256,0,st>>>(L[5], L[7], veff, din);
  k_matvec<<<CDIV(NQ,8),256,0,st>>>(qin, din, veff, sdq, NQ);    // cq dst=q
  // ---- aggregate into c (cc + qc share one agg buffer; reference sums them)
  fillz(st, aggc, NC * DD);
  fillz(st, menc, NC); fillz(st, den, NC);
  k_edge_max<<<CDIV(ECCN,256),256,0,st>>>(ccS, ccD, sscc, sdccc, ebuf, menc, ECCN);
  k_edge_exp<<<CDIV(ECCN,256),256,0,st>>>(ccD, ebuf, menc, den, ECCN);
  k_edge_agg<<<CDIV(ECCN*32,256),256,0,st>>>(ccS, ccD, ebuf, den, cout, 384, aggc, ECCN);
  fillz(st, menc, NC); fillz(st, den, NC);
  k_edge_max<<<CDIV(EQCN,256),256,0,st>>>(qcS, qcD, ssq, sdcqc, ebuf, menc, EQCN);
  k_edge_exp<<<CDIV(EQCN,256),256,0,st>>>(qcD, ebuf, menc, den, EQCN);
  k_edge_agg<<<CDIV(EQCN*32,256),256,0,st>>>(qcS, qcD, ebuf, den, qout, 256, aggc, EQCN);
  // ---- aggregate into q (cq)
  fillz(st, aggq, NQ * DD);
  fillz(st, menc, NQ); fillz(st, den, NQ);
  k_edge_max<<<CDIV(ECQN,256),256,0,st>>>(cqS, cqD, sscq, sdq, ebuf, menc, ECQN);
  k_edge_exp<<<CDIV(ECQN,256),256,0,st>>>(cqD, ebuf, menc, den, ECQN);
  k_edge_agg<<<CDIV(ECQN*32,256),256,0,st>>>(cqS, cqD, ebuf, den, cout + DD, 384, aggq, ECQN);
  // ---- combine + relu
  k_combine_relu<<<CDIV(NC*DD,256),256,0,st>>>(c_out, aggc, cout, 384, 2*DD,
                                               L[4], L[18], L[11], NC);
  k_combine_relu<<<CDIV(NQ*DD,256),256,0,st>>>(q_out, aggq, qout, 256, DD,
                                               L[9], L[13], nullptr, NQ);
}

extern "C" void kernel_launch(void* const* d_in, const int* in_sizes, int n_in,
                              void* d_out, int out_size, void* d_ws, size_t ws_size,
                              hipStream_t stream) {
  (void)in_sizes; (void)n_in; (void)out_size; (void)ws_size;
  // d_in layout (params sorted-key pytree, then remaining inputs):
  // 0 c_x | 1 gru_Whh | 2 gru_Wih | 3 gru_bhh | 4 gru_bih | 5..23 layer0 |
  // 24..42 layer1 | 43 p1_W 44 p1_b 45 p2_W 46 p2_b | 47 q_x |
  // 48 cc_edge 49 qc_edge 50 cq_edge 51 q_seq 52 concepts 53 operate
  const float* c_x = (const float*)d_in[0];
  const float* q_x = (const float*)d_in[47];
  const float* L0[19]; const float* L1[19];
  for (int i = 0; i < 19; ++i) { L0[i] = (const float*)d_in[5 + i];
                                 L1[i] = (const float*)d_in[24 + i]; }
  const int* ccE = (const int*)d_in[48];
  const int* qcE = (const int*)d_in[49];
  const int* cqE = (const int*)d_in[50];
  const int* q_seq    = (const int*)d_in[51];
  const int* concepts = (const int*)d_in[52];
  const float* operate = (const float*)d_in[53];

  // -------- workspace arenas (aliased across phases) --------
  char* base = (char*)d_ws; size_t off = 0;
  auto A = [&](size_t b) { char* p = base + off; off = (off + b + 255) & ~(size_t)255; return p; };
  char* arenaA = A(76800000);  // qbf; later: gi | allh | pxbf | hidpre
  char* arenaB = A(51200000);  // qout; later: eq | ec | xbf
  float* q1   = (float*)A((size_t)NQ * DD * 4);
  u16*   q1bf = (u16*)  A((size_t)NQ * DD * 2);
  float* aggq = (float*)A((size_t)NQ * DD * 4);
  u16*   cbf  = (u16*)  A((size_t)NCP * BERT * 2);
  float* cout = (float*)A((size_t)NCP * 384 * 4);
  float* c1   = (float*)A((size_t)NCP * DD * 4);
  u16*   c1bf = (u16*)  A((size_t)NCP * DD * 2);
  float* aggc = (float*)A((size_t)NC * DD * 4);
  u16* qpack = (u16*)A(256 * 768 * 2);
  u16* cpack = (u16*)A(384 * 768 * 2);
  u16* wihbf = (u16*)A(GG * HH * 2);
  u16* whhbf = (u16*)A(GG * HH * 2);
  u16* p1bf  = (u16*)A(DD * 512 * 2);
  float* ssq   = (float*)A(NQ * 4);
  float* sdq   = (float*)A(NQ * 4);
  float* sscc  = (float*)A(4096);
  float* sscq  = (float*)A(4096);
  float* sdccc = (float*)A(4096);
  float* sdcqc = (float*)A(4096);
  u32*   menc  = (u32*)  A(NQ * 4);
  float* den   = (float*)A(NQ * 4);
  float* ebuf  = (float*)A(ECQN * 4);
  float* veff  = (float*)A(BERT * 4);
  // aliases
  u16*   qbf    = (u16*)arenaA;
  float* gi     = (float*)arenaA;                    // after layer GEMMs done
  float* allh   = (float*)(arenaA + 39321600);
  u16*   pxbf   = (u16*)  (arenaA + 52428800);
  float* hidpre = (float*)(arenaA + 65536000);
  float* qout   = (float*)arenaB;
  float* eq     = (float*)arenaB;                    // after layer 1 done
  float* ecv    = (float*)(arenaB + 6553600);
  u16*   xbf    = (u16*)  (arenaB + 13107200);
  float* q2 = q1;  // layer-1 combine writes in place (inputs are aggq/qout)
  float* c2 = c1;

  // -------- feature conversions --------
  conv(stream, qbf, q_x, NQ * BERT);
  fillz(stream, cbf, NCP * BERT / 2);
  conv(stream, cbf, c_x, NC * BERT);

  // -------- layer 0 (din = 768) --------
  run_layer(stream, BERT, L0, q_x, qbf, c_x, cbf,
            ccE, ccE + ECCN, qcE, qcE + EQCN, cqE, cqE + ECQN,
            qpack, cpack, qout, cout, ssq, sdq, sscc, sscq, sdccc, sdcqc,
            menc, den, ebuf, veff, aggq, aggc, q1, c1);

  // -------- layer 1 (din = 128) --------
  conv(stream, q1bf, q1, NQ * DD);
  fillz(stream, c1bf, NCP * DD / 2);
  conv(stream, c1bf, c1, NC * DD);
  run_layer(stream, DD, L1, q1, q1bf, c1, c1bf,
            ccE, ccE + ECCN, qcE, qcE + EQCN, cqE, cqE + ECQN,
            qpack, cpack, qout, cout, ssq, sdq, sscc, sscq, sdccc, sdcqc,
            menc, den, ebuf, veff, aggq, aggc, q2, c2);

  // -------- sequence phase --------
  k_embed_q<<<CDIV(BL*DD,256),256,0,stream>>>(q_seq, q2, eq);
  k_embed_c<<<CDIV(BL*DD,256),256,0,stream>>>(concepts, c2, ecv);
  k_pack_x <<<CDIV(BL*DD,256),256,0,stream>>>(ecv, operate, xbf);
  conv(stream, wihbf, (const float*)d_in[2], GG * HH);
  conv(stream, whhbf, (const float*)d_in[1], GG * HH);
  conv(stream, p1bf,  (const float*)d_in[43], DD * 512);
  gemm(stream, xbf, wihbf, gi, BL / 16, 12, HH);     // gi = x @ Wih^T (no bias)
  size_t gru_lds = 64 * 264 * 2 + 64 * 776 * 4;      // 232448 B dynamic LDS
  k_gru<<<1, 512, gru_lds, stream>>>(whhbf, gi,
                                     (const float*)d_in[4],   // bih
                                     (const float*)d_in[3],   // bhh
                                     allh);
  k_pack_px<<<CDIV(BL*512,256),256,0,stream>>>(allh, eq, ecv, pxbf);
  gemm(stream, pxbf, p1bf, hidpre, BL / 16, 2, 512); // px @ p1_W^T
  k_head<<<CDIV(BL,8),256,0,stream>>>(hidpre,
                                      (const float*)d_in[44],  // p1_b
                                      (const float*)d_in[45],  // p2_W
                                      (const float*)d_in[46],  // p2_b
                                      (float*)d_out, BL);
}
